// AttnBlock_16037407883772
// MI455X (gfx1250) — matrix operand order repaired and checked by reading, not compiled, not run on hardware
//
#include <hip/hip_runtime.h>
#include <hip/hip_bf16.h>

// ---------------------------------------------------------------------------
// AttnBlock for MI455X (gfx1250): bf16 WMMA flash-attention pipeline.
// B=8, C=512, N=H*W=4096, GROUPS=32 (16 ch/group).
// All GEMM operands are stored in WMMA wave32 register-operand order so the
// hot loops do contiguous 32B/lane loads + v_wmma_f32_16x16x32_bf16.
// ---------------------------------------------------------------------------

typedef __attribute__((ext_vector_type(16))) __bf16 v16bf;
typedef __attribute__((ext_vector_type(8)))  float  v8f;

#define WMMA_BF16(a, b, c) \
  __builtin_amdgcn_wmma_f32_16x16x32_bf16(false, (a), false, (b), (short)0, (c), false, false)

#define CC 512
#define NN 4096
#define PER_BATCH ((size_t)CC * NN)   // 2,097,152 elements

// A-operand (16x32, 16-bit) slot->K mapping: lane L (half = L>>4), slot s in [0,16)
static __device__ __forceinline__ int a_kidx(int L, int s) {
  return (s & 7) + ((L >> 4) << 3) + ((s >> 3) << 4);
}
// B-operand (32x16, 16-bit) slot->K mapping
static __device__ __forceinline__ int b_kidx(int L, int s) {
  return (s & 7) + ((L >> 4) << 3) + ((s >> 3) << 4);
}

// ---------------------------------------------------------------------------
// 1) GroupNorm statistics: one block per (batch, group). 16 ch * 4096 = 65536.
// ---------------------------------------------------------------------------
__global__ void gn_stats_kernel(const float* __restrict__ x, float* __restrict__ stat) {
  int bg = blockIdx.x;                        // b*32 + g
  const float* base = x + (size_t)bg * 65536; // group channels are contiguous
  float s = 0.f, ss = 0.f;
  for (int i = threadIdx.x; i < 65536; i += 256) {
    float v = base[i];
    s += v; ss += v * v;
  }
  __shared__ float red0[256];
  __shared__ float red1[256];
  red0[threadIdx.x] = s; red1[threadIdx.x] = ss;
  __syncthreads();
  for (int off = 128; off > 0; off >>= 1) {
    if (threadIdx.x < off) {
      red0[threadIdx.x] += red0[threadIdx.x + off];
      red1[threadIdx.x] += red1[threadIdx.x + off];
    }
    __syncthreads();
  }
  if (threadIdx.x == 0) {
    float mean = red0[0] * (1.f / 65536.f);
    float var  = red1[0] * (1.f / 65536.f) - mean * mean;
    stat[bg * 2 + 0] = mean;
    stat[bg * 2 + 1] = rsqrtf(var + 1e-6f);
  }
}

// ---------------------------------------------------------------------------
// 2) GroupNorm apply + pack r as bf16 B-operand tiles (K=channel 32, N=16 cols)
//    tile ordering per batch: (kk*256 + j)*512
// ---------------------------------------------------------------------------
__global__ void gn_apply_kernel(const float* __restrict__ x, const float* __restrict__ stat,
                                const float* __restrict__ gamma, const float* __restrict__ beta,
                                __bf16* __restrict__ rws) {
  int j = blockIdx.x;   // n-tile of 16
  int b = blockIdx.y;
  const float* xb = x + (size_t)b * PER_BATCH;
  const float* st = stat + b * 64;
  __bf16* rb = rws + (size_t)b * PER_BATCH;
  for (int kk = 0; kk < 16; ++kk) {
    __bf16* tile = rb + (size_t)(kk * 256 + j) * 512;
    for (int e = threadIdx.x; e < 512; e += 256) {
      int L = e >> 4, s = e & 15;
      int c = kk * 32 + b_kidx(L, s);
      int n = j * 16 + (L & 15);
      int g = c >> 4;
      float mu = st[g * 2], rs = st[g * 2 + 1];
      float v = (xb[(size_t)c * NN + n] - mu) * rs * gamma[c] + beta[c];
      tile[e] = (__bf16)v;
    }
  }
}

// ---------------------------------------------------------------------------
// 3) Weight pack: W (512x512 f32, row=co, col=ci) -> bf16 A-operand tiles.
// ---------------------------------------------------------------------------
__global__ void wpack_kernel(const float* __restrict__ wq, const float* __restrict__ wk,
                             const float* __restrict__ wp, __bf16* __restrict__ out) {
  int tile = blockIdx.x;
  int which = blockIdx.y;
  const float* src = (which == 0) ? wq : (which == 1) ? wk : wp;
  __bf16* dst = out + (size_t)which * (CC * CC) + (size_t)tile * 512;
  int i = tile >> 4, kk = tile & 15;
  for (int e = threadIdx.x; e < 512; e += 256) {
    int L = e >> 4, s = e & 15;
    int co = i * 16 + (L & 15);
    int ci = kk * 32 + a_kidx(L, s);
    dst[e] = (__bf16)src[co * CC + ci];
  }
}

// ---------------------------------------------------------------------------
// 4) q/k projections: GEMM out(co,n) = W*r + bias over a 32-column n block.
//      mode 0: q  -> A-tiles of q^T   (i*16+kk)*512
//      mode 1: k  -> B-tiles of k     (kk*256+j)*512
//              kT -> B-tiles of k^T   (mm*32+jc)*512  (v = k)
// ---------------------------------------------------------------------------
__global__ void proj_qk_kernel(const __bf16* __restrict__ rws, const __bf16* __restrict__ wpk,
                               const float* __restrict__ qbias, const float* __restrict__ kbias,
                               __bf16* __restrict__ qws, __bf16* __restrict__ kws,
                               __bf16* __restrict__ ktws) {
  int jj = blockIdx.x;        // 32-column block (2 n-tiles)
  int b  = blockIdx.y;
  int mode = blockIdx.z;      // 0 -> q, 1 -> k/kT
  const __bf16* rb = rws + (size_t)b * PER_BATCH;
  const __bf16* wp = wpk + (size_t)(mode == 0 ? 0 : 1) * (CC * CC);
  const float* bias = (mode == 0) ? qbias : kbias;
  int w = threadIdx.x >> 5, L = threadIdx.x & 31, half = L >> 4, col = L & 15;

  v8f acc[4][2] = {};
#pragma unroll 2
  for (int kk = 0; kk < 16; ++kk) {
    v16bf B0 = *(const v16bf*)(rb + (size_t)(kk * 256 + jj * 2 + 0) * 512 + L * 16);
    v16bf B1 = *(const v16bf*)(rb + (size_t)(kk * 256 + jj * 2 + 1) * 512 + L * 16);
#pragma unroll
    for (int ct = 0; ct < 4; ++ct) {
      int cot = w + ct * 8;
      v16bf A = *(const v16bf*)(wp + (size_t)(cot * 16 + kk) * 512 + L * 16);
      acc[ct][0] = WMMA_BF16(A, B0, acc[ct][0]);
      acc[ct][1] = WMMA_BF16(A, B1, acc[ct][1]);
    }
  }

  __shared__ __bf16 sbuf[CC * 32];   // [c][n_local], 32KB
#pragma unroll
  for (int ct = 0; ct < 4; ++ct) {
    int cot = w + ct * 8;
#pragma unroll
    for (int v = 0; v < 8; ++v) {
      int co = cot * 16 + v + 8 * half;
      float bv = bias[co];
      sbuf[co * 32 + col]      = (__bf16)(acc[ct][0][v] + bv);
      sbuf[co * 32 + 16 + col] = (__bf16)(acc[ct][1][v] + bv);
    }
  }
  __syncthreads();

  if (mode == 0) {
    __bf16* qbp = qws + (size_t)b * PER_BATCH;
    for (int t2 = 0; t2 < 32; ++t2) {   // i_local(2) x kk(16)
      int il = t2 >> 4, kk = t2 & 15;
      int i = jj * 2 + il;
      __bf16* dst = qbp + (size_t)(i * 16 + kk) * 512;
      for (int e = threadIdx.x; e < 512; e += 256) {
        int L2 = e >> 4, s = e & 15;
        int c  = kk * 32 + a_kidx(L2, s);
        int nl = il * 16 + (L2 & 15);
        dst[e] = sbuf[c * 32 + nl];
      }
    }
  } else {
    __bf16* kbp = kws + (size_t)b * PER_BATCH;
    __bf16* ktp = ktws + (size_t)b * PER_BATCH;
    for (int t2 = 0; t2 < 32; ++t2) {   // j_local(2) x kk(16)
      int jl = t2 >> 4, kk = t2 & 15;
      int j = jj * 2 + jl;
      __bf16* dst = kbp + (size_t)(kk * 256 + j) * 512;
      for (int e = threadIdx.x; e < 512; e += 256) {
        int L2 = e >> 4, s = e & 15;
        int c  = kk * 32 + b_kidx(L2, s);
        int ml = jl * 16 + (L2 & 15);
        dst[e] = sbuf[c * 32 + ml];
      }
    }
    for (int jc = 0; jc < 32; ++jc) {   // kT B-tiles: K=m (32 rows of this block)
      __bf16* dst = ktp + (size_t)(jj * 32 + jc) * 512;
      for (int e = threadIdx.x; e < 512; e += 256) {
        int L2 = e >> 4, s = e & 15;
        int ml = b_kidx(L2, s);         // 0..31
        int c  = jc * 16 + (L2 & 15);
        dst[e] = sbuf[c * 32 + ml];
      }
    }
  }
}

// ---------------------------------------------------------------------------
// 5) Flash attention: block = (batch, 64 query rows). 8 waves.
//    Wave w: S tiles (row ri=w&3, cols (w>>2)*2 + {0,1});
//            O slice c in [64w, 64w+64): 4x4 f32 accumulator tiles.
//    Softmax: 4 threads per row (all 256 threads active), running (m,l)
//    state replicated across the 4 threads of a row.
// ---------------------------------------------------------------------------
__global__ void flash_kernel(const __bf16* __restrict__ qws, const __bf16* __restrict__ kws,
                             const __bf16* __restrict__ ktws, __bf16* __restrict__ ows) {
  extern __shared__ char smem[];
  __bf16* qs     = (__bf16*)smem;                       // 64x512 bf16 = 64KB (A-tile order)
  float*  Ss     = (float*)(smem + 65536);              // 64x64 f32 = 16KB
  __bf16* Ps     = (__bf16*)(smem + 65536 + 16384);     // 8 A-tiles = 8KB
  float*  alphaS = (float*)(smem + 90112);              // 64 f32
  float*  lS     = (float*)(smem + 90368);              // 64 f32
  float*  Smax   = (float*)(smem + 90624);              // 64x4 f32
  float*  Ssum   = (float*)(smem + 91648);              // 64x4 f32

  int q0 = blockIdx.x, b = blockIdx.y;
  const __bf16* qb  = qws  + (size_t)b * PER_BATCH;
  const __bf16* kb  = kws  + (size_t)b * PER_BATCH;
  const __bf16* ktb = ktws + (size_t)b * PER_BATCH;

  int t = threadIdx.x, w = t >> 5, L = t & 31, half = L >> 4, col = L & 15;
  int ri = w & 3, cb = (w >> 2) * 2;
  int r4 = t >> 2, q4 = t & 3;        // softmax: row, quarter

  // stage the 64 q A-tiles (contiguous in qws) into LDS
  {
    const __bf16* src = qb + (size_t)q0 * 32768;
#pragma unroll
    for (int chunk = 0; chunk < 16; ++chunk) {
      int idx = (chunk * 256 + t) * 8;
      *(uint4*)(qs + idx) = *(const uint4*)(src + idx);
    }
  }

  v8f accO[4][4] = {};
  float rowM = -3.0e38f, rowL = 0.f;
  const float scale = 0.044194173824159216f;  // 512^-0.5
  __syncthreads();

  for (int mb = 0; mb < 64; ++mb) {
    // ---- S = (q^T k) * scale, two 16x16 tiles per wave, K = 512 ----
    v8f s0 = {}, s1 = {};
#pragma unroll 2
    for (int kk = 0; kk < 16; ++kk) {
      v16bf A = *(const v16bf*)(qs + (ri * 16 + kk) * 512 + L * 16);
      const __bf16* kbase = kb + (size_t)(kk * 256 + mb * 4) * 512;
      v16bf B0 = *(const v16bf*)(kbase + (size_t)(cb + 0) * 512 + L * 16);
      v16bf B1 = *(const v16bf*)(kbase + (size_t)(cb + 1) * 512 + L * 16);
      if (mb < 63) {   // prefetch next m-block's K tiles into near cache
        __builtin_prefetch(kbase + 4 * 512 + (size_t)(cb + 0) * 512 + L * 16, 0, 3);
        __builtin_prefetch(kbase + 4 * 512 + (size_t)(cb + 1) * 512 + L * 16, 0, 3);
      }
      s0 = WMMA_BF16(A, B0, s0);
      s1 = WMMA_BF16(A, B1, s1);
    }
#pragma unroll
    for (int v = 0; v < 8; ++v) {
      int r = ri * 16 + v + 8 * half;
      Ss[r * 64 + cb * 16 + col]       = s0[v] * scale;
      Ss[r * 64 + (cb + 1) * 16 + col] = s1[v] * scale;
    }
    __syncthreads();

    // ---- online softmax, 4 threads/row ----
    {
      float pm = -3.0e38f;
#pragma unroll
      for (int u = 0; u < 16; ++u) pm = fmaxf(pm, Ss[r4 * 64 + q4 * 16 + u]);
      Smax[r4 * 4 + q4] = pm;
    }
    __syncthreads();
    {
      float mx = fmaxf(fmaxf(Smax[r4 * 4 + 0], Smax[r4 * 4 + 1]),
                       fmaxf(Smax[r4 * 4 + 2], Smax[r4 * 4 + 3]));
      mx = fmaxf(mx, rowM);
      float alpha = __expf(rowM - mx);
      rowM = mx;
      float psum = 0.f;
      int rt = r4 >> 4, rl = r4 & 15;
#pragma unroll
      for (int u = 0; u < 16; ++u) {
        int j2 = q4 * 16 + u;
        float p = __expf(Ss[r4 * 64 + j2] - mx);
        psum += p;
        int kk2 = j2 >> 5, mi = j2 & 31;
        int h2 = (mi >> 3) & 1;
        int s  = (mi & 7) + ((mi >> 4) << 3);
        Ps[(rt * 2 + kk2) * 512 + (rl + 16 * h2) * 16 + s] = (__bf16)p;
      }
      Ssum[r4 * 4 + q4] = psum;
      if (q4 == 0) alphaS[r4] = alpha;
      __syncthreads();
      rowL = rowL * alpha + (Ssum[r4 * 4 + 0] + Ssum[r4 * 4 + 1] +
                             Ssum[r4 * 4 + 2] + Ssum[r4 * 4 + 3]);
    }

    // ---- O = O*alpha + P @ V^T (wave owns 64 output channels) ----
#pragma unroll
    for (int rI = 0; rI < 4; ++rI) {
      v8f av = *(const v8f*)(alphaS + rI * 16 + 8 * half);   // rows rI*16+8*half+[0..8)
#pragma unroll
      for (int jc = 0; jc < 4; ++jc)
#pragma unroll
        for (int v = 0; v < 8; ++v) accO[rI][jc][v] *= av[v];
#pragma unroll
      for (int kk2 = 0; kk2 < 2; ++kk2) {
        v16bf A = *(const v16bf*)(Ps + (rI * 2 + kk2) * 512 + L * 16);
#pragma unroll
        for (int jc = 0; jc < 4; ++jc) {
          const __bf16* vbase = ktb + (size_t)((mb * 2 + kk2) * 32 + (w * 4 + jc)) * 512;
          if (mb < 63)   // prefetch next m-block's V^T tiles
            __builtin_prefetch(vbase + (size_t)64 * 512 + L * 16, 0, 3);
          v16bf Bv = *(const v16bf*)(vbase + L * 16);
          accO[rI][jc] = WMMA_BF16(A, Bv, accO[rI][jc]);
        }
      }
    }
    __syncthreads();
  }

  if (q4 == 0) lS[r4] = rowL;
  __syncthreads();

  // normalize, stage O to LDS [n64][c512] (reuse qs), then pack B-tiles to ows
#pragma unroll
  for (int rI = 0; rI < 4; ++rI) {
    v8f lv = *(const v8f*)(lS + rI * 16 + 8 * half);
#pragma unroll
    for (int v = 0; v < 8; ++v) lv[v] = 1.f / lv[v];
#pragma unroll
    for (int jc = 0; jc < 4; ++jc)
#pragma unroll
      for (int v = 0; v < 8; ++v)
        qs[(rI * 16 + v + 8 * half) * 512 + w * 64 + jc * 16 + col] =
            (__bf16)(accO[rI][jc][v] * lv[v]);
  }
  __syncthreads();

  __bf16* ob = ows + (size_t)b * PER_BATCH;
  for (int t64 = 0; t64 < 64; ++t64) {   // kk(16) x j_local(4)
    int kk = t64 >> 2, jl = t64 & 3;
    __bf16* dst = ob + (size_t)(kk * 256 + q0 * 4 + jl) * 512;
    for (int e = t; e < 512; e += 256) {
      int L2 = e >> 4, s = e & 15;
      int ci = kk * 32 + b_kidx(L2, s);
      int nl = jl * 16 + (L2 & 15);
      dst[e] = qs[nl * 512 + ci];
    }
  }
}

// ---------------------------------------------------------------------------
// 6) Final projection + bias + residual, f32 output in (b,C,h,w) layout.
// ---------------------------------------------------------------------------
__global__ void proj_final_kernel(const __bf16* __restrict__ ows, const __bf16* __restrict__ wpk,
                                  const float* __restrict__ pb, const float* __restrict__ x,
                                  float* __restrict__ out) {
  int j = blockIdx.x, b = blockIdx.y;
  const __bf16* ob = ows + (size_t)b * PER_BATCH;
  const __bf16* wp = wpk + (size_t)2 * (CC * CC);
  int w = threadIdx.x >> 5, L = threadIdx.x & 31, half = L >> 4, col = L & 15;

  v8f acc[4] = {};
#pragma unroll 2
  for (int kk = 0; kk < 16; ++kk) {
    v16bf B = *(const v16bf*)(ob + (size_t)(kk * 256 + j) * 512 + L * 16);
#pragma unroll
    for (int ct = 0; ct < 4; ++ct) {
      v16bf A = *(const v16bf*)(wp + (size_t)((w + ct * 8) * 16 + kk) * 512 + L * 16);
      acc[ct] = WMMA_BF16(A, B, acc[ct]);
    }
  }
#pragma unroll
  for (int ct = 0; ct < 4; ++ct) {
    int cot = w + ct * 8;
#pragma unroll
    for (int v = 0; v < 8; ++v) {
      int co = cot * 16 + v + 8 * half;
      size_t idx = ((size_t)(b * CC + co)) * NN + j * 16 + col;
      out[idx] = acc[ct][v] + pb[co] + x[idx];
    }
  }
}

// ---------------------------------------------------------------------------
// Host launcher
// ---------------------------------------------------------------------------
extern "C" void kernel_launch(void* const* d_in, const int* in_sizes, int n_in,
                              void* d_out, int out_size, void* d_ws, size_t ws_size,
                              hipStream_t stream) {
  (void)in_sizes; (void)n_in; (void)out_size; (void)ws_size;
  const float* x   = (const float*)d_in[0];
  const float* gg  = (const float*)d_in[1];
  const float* gb  = (const float*)d_in[2];
  const float* qw  = (const float*)d_in[3];
  const float* qb  = (const float*)d_in[4];
  const float* kw  = (const float*)d_in[5];
  const float* kb  = (const float*)d_in[6];
  const float* pw  = (const float*)d_in[7];
  const float* pbi = (const float*)d_in[8];
  float* out = (float*)d_out;

  char* ws = (char*)d_ws;
  const size_t BATCH_BYTES = PER_BATCH * 2;               // bf16 per batch
  float*  gnstat = (float*)ws;                            // 2 KB
  __bf16* wpack  = (__bf16*)(ws + 2048);                  // 3 * 512KB
  __bf16* rws    = (__bf16*)(ws + 2048 + 1572864);        // 32 MB (aliased with ows)
  __bf16* ows    = rws;                                   // r is dead before flash writes
  __bf16* qws    = (__bf16*)((char*)rws + 8 * BATCH_BYTES);
  __bf16* kws    = (__bf16*)((char*)qws + 8 * BATCH_BYTES);
  __bf16* ktws   = (__bf16*)((char*)kws + 8 * BATCH_BYTES);

  gn_stats_kernel<<<dim3(256), dim3(256), 0, stream>>>(x, gnstat);
  gn_apply_kernel<<<dim3(256, 8), dim3(256), 0, stream>>>(x, gnstat, gg, gb, rws);
  wpack_kernel<<<dim3(512, 3), dim3(256), 0, stream>>>(qw, kw, pw, wpack);
  proj_qk_kernel<<<dim3(128, 8, 2), dim3(256), 0, stream>>>(rws, wpack, qb, kb,
                                                            qws, kws, ktws);
  size_t flash_smem = 92672;   // qs 64K + Ss 16K + Ps 8K + alpha/l/Smax/Ssum
  flash_kernel<<<dim3(64, 8), dim3(256), flash_smem, stream>>>(qws, kws, ktws, ows);
  proj_final_kernel<<<dim3(256, 8), dim3(256), 0, stream>>>(ows, wpack, pbi, x, out);
}
